// BidirectionalGRUEncoder_14877766714157
// MI455X (gfx1250) — matrix operand order repaired and checked by reading, not compiled, not run on hardware
//
#include <hip/hip_runtime.h>
#include <hip/hip_bf16.h>

// ---------------- problem constants ----------------
#define BB      32
#define SS      384
#define DCTX    300
#define HH      600
#define INR     303           // real input dim
#define ROWS    (BB*SS)       // 12288 scan steps / emb rows
#define NKT     10            // K tiles of 32 (303 -> 320)
#define NOUT    1800          // 3*H gi columns
#define NNT2    114           // N tiles of 16, padded even for 2-wide N blocking
#define NPAD    (NNT2*16)     // 1824 (gi row stride)
#define NMT     (ROWS/16)     // 768 M tiles
#define LDH     1216          // all_h row stride (floats): 4864 B = 38 * 128B lines exactly
#define LDW     608           // LDS W_hh row stride (floats, 152 float4s, zero padded)
#define NWG     25            // workgroups cooperating per scan direction
#define IH      24            // h indices per workgroup (25*24 = 600)
#define FRAG    512           // __bf16 elements per 32-lane fragment tile (32 lanes * 16)

typedef __attribute__((ext_vector_type(16))) __bf16 v16bf;
typedef __attribute__((ext_vector_type(8)))  float  v8f;

// ---------------- init: zero barrier counters ----------------
__global__ void k_init(unsigned int* counters) {
  if (threadIdx.x < 2 && blockIdx.x == 0) counters[threadIdx.x] = 0u;
}

// ---------------- prep A: emb -> bf16 WMMA A-fragments ----------------
// A fragment layout (16-bit A 16x32): lane<16: row=lane, K = 0..15;
// lane>=16: row=lane-16, K = 16..31.  Stored contiguously per (tile,lane).
__global__ void k_prep_A(const float* __restrict__ ctx, const int* __restrict__ tags,
                         const float* __restrict__ bio, __bf16* __restrict__ Asw) {
  int gid = blockIdx.x * 256 + threadIdx.x;        // one thread per (mt,kt,lane)
  if (gid >= NMT * NKT * 32) return;
  int lane = gid & 31;
  int tile = gid >> 5;
  int kt = tile % NKT;
  int mt = tile / NKT;
  int row = mt * 16 + (lane & 15);
  int hi  = lane >> 4;
  int tag = tags[row];
  v16bf frag;
#pragma unroll
  for (int e = 0; e < 16; ++e) {
    int k = kt * 32 + hi * 16 + e;
    float v;
    if (k < 3)          v = bio[tag * 3 + k];
    else if (k < INR)   v = ctx[(size_t)row * DCTX + (k - 3)];
    else                v = 0.0f;
    frag[e] = (__bf16)v;
  }
  *(v16bf*)(Asw + (size_t)gid * 16) = frag;
}

// ---------------- prep B: W_ih^T -> bf16 WMMA B-fragments ----------------
// B fragment layout (16-bit B 32x16): lane<16: col=lane, K=0..15; lane>=16: col=lane-16, K=16..31.
__global__ void k_prep_B(const float* __restrict__ W_ih, __bf16* __restrict__ Bsw) {
  int gid = blockIdx.x * 256 + threadIdx.x;        // one thread per (kt,nt,lane)
  if (gid >= NKT * NNT2 * 32) return;
  int lane = gid & 31;
  int tile = gid >> 5;
  int nt = tile % NNT2;
  int kt = tile / NNT2;
  int col   = nt * 16 + (lane & 15);
  int kbase = kt * 32 + (lane >> 4) * 16;
  v16bf frag;
#pragma unroll
  for (int e = 0; e < 16; ++e) {
    int k = kbase + e;
    float v = (k < INR && col < NOUT) ? W_ih[(size_t)col * INR + k] : 0.0f;
    frag[e] = (__bf16)v;
  }
  *(v16bf*)(Bsw + (size_t)gid * 16) = frag;
}

// ---------------- WMMA GEMM: gi = embA @ W_ihT + b_ih ----------------
// Register-blocked 4 M-tiles x 2 N-tiles per wave (8 WMMAs / 6 fragment loads per K step).
// All fragment addresses are base-pointer + compile-time immediate offsets (fits 24-bit
// instruction offset), so the inner loop has no per-load address arithmetic and no spills.
__global__ void k_gemm(const __bf16* __restrict__ Asw, const __bf16* __restrict__ Bsw,
                       const float* __restrict__ b_ih, float* __restrict__ gi) {
  int wave = (int)(threadIdx.x >> 5);
  int lane = (int)(threadIdx.x & 31);
  int tile = blockIdx.x * 8 + wave;                // (768/4)*(114/2) = 10944 groups, grid = 1368
  if (tile >= (NMT / 4) * (NNT2 / 2)) return;      // wave-uniform: EXEC stays all-ones
  int ntb = tile % (NNT2 / 2);
  int mtb = tile / (NNT2 / 2);
  int mt0 = mtb * 4;
  int nt0 = ntb * 2;
  // base pointers; per-(im,kt)/(kt,in) displacements are constants after unrolling
  const __bf16* A0 = Asw + ((size_t)(mt0 * NKT) * 32 + lane) * 16;
  const __bf16* B0 = Bsw + ((size_t)(nt0) * 32 + lane) * 16;
  v8f acc[4][2] = {};
#pragma unroll 2
  for (int kt = 0; kt < NKT; ++kt) {
    v16bf a[4], b[2];
#pragma unroll
    for (int im = 0; im < 4; ++im)
      a[im] = *(const v16bf*)(A0 + (im * NKT + kt) * FRAG);       // offset = im*10240B + kt*1024B
#pragma unroll
    for (int in = 0; in < 2; ++in)
      b[in] = *(const v16bf*)(B0 + (kt * NNT2 + in) * FRAG);      // offset = kt*116736B + in*1024B
#pragma unroll
    for (int im = 0; im < 4; ++im)
#pragma unroll
      for (int in = 0; in < 2; ++in)
        acc[im][in] = __builtin_amdgcn_wmma_f32_16x16x32_bf16(
            false, a[im], false, b[in], (short)0, acc[im][in], false, false);
  }
  // C/D layout: lane<16: N=lane, M=j ; lane>=16: N=lane-16, M=j+8
#pragma unroll
  for (int in = 0; in < 2; ++in) {
    int col = (nt0 + in) * 16 + (lane & 15);
    if (col < NOUT) {
      float bias = b_ih[col];
#pragma unroll
      for (int im = 0; im < 4; ++im) {
        int rbase = (mt0 + im) * 16 + (lane >> 4) * 8;
#pragma unroll
        for (int j = 0; j < 8; ++j)
          gi[(size_t)(rbase + j) * NPAD + col] = acc[im][in][j] + bias;
      }
    }
  }
}

// ---------------- cooperative sequential GRU scan ----------------
// 2 directions * 25 WGs. WG handles h-indices [i0, i0+24); its 72 W_hh rows live in LDS
// (~178 KB fp32 -- fits CDNA5's 320 KB WGP LDS). float4 ds_load_b128 dot products keep
// the latency-critical matvec at VALU rate instead of LDS-b32 rate. Per-step cross-WG
// sync via monotonic atomic counter; h_t flows through all_h rows (exact-cacheline stride).
__global__ void k_scan(const float* __restrict__ gi, const float* __restrict__ W_hh,
                       const float* __restrict__ b_hh, float* __restrict__ all_h,
                       unsigned int* __restrict__ counters) {
  extern __shared__ float smem[];
  float* Wlds = smem;                  // 72 * LDW
  float* hbuf = smem + 72 * LDW;       // LDW floats (600 real + zero pad)
  float* ghl  = hbuf + LDW;            // 72
  int dir = blockIdx.x / NWG;
  int wg  = blockIdx.x % NWG;
  int tid = (int)threadIdx.x;
  int i0  = wg * IH;

  // Stage this WG's W_hh rows {i, 600+i, 1200+i} into LDS (once), zero padded to LDW.
  for (int idx = tid; idx < 72 * LDW; idx += 256) {
    int lr = idx / LDW, k = idx % LDW;
    int c = lr / IH, ii = lr % IH;
    Wlds[idx] = (k < HH) ? W_hh[(size_t)(c * HH + i0 + ii) * HH + k] : 0.0f;
  }
  float bh0 = 0.f, bh1 = 0.f, bh2 = 0.f;
  if (tid < IH) {
    bh0 = b_hh[i0 + tid];
    bh1 = b_hh[HH + i0 + tid];
    bh2 = b_hh[2 * HH + i0 + tid];
  }
  __syncthreads();

  int wid = tid >> 5, lane = tid & 31;
  float4* hb4 = (float4*)hbuf;
  unsigned int target = 0;
  for (int t = 0; t < ROWS; ++t) {
    // load h_{t-1} (h_0 = 0) as float4; 150 real float4s + 2 zero pad
    for (int c = tid; c < LDW / 4; c += 256) {
      float4 v = make_float4(0.f, 0.f, 0.f, 0.f);
      if (t > 0 && c < HH / 4)
        v = *(const float4*)(all_h + (size_t)(t - 1) * LDH + dir * HH + c * 4);
      hb4[c] = v;
    }
    __syncthreads();

    // gh = W_slice @ h : 8 waves x 9 rows, float4 lane-strided dot + wave reduction
#pragma unroll
    for (int q = 0; q < 9; ++q) {
      int lr = wid * 9 + q;
      const float4* wr4 = (const float4*)(Wlds + lr * LDW);
      float p = 0.0f;
      for (int c = lane; c < LDW / 4; c += 32) {
        float4 w = wr4[c];
        float4 h = hb4[c];
        p += w.x * h.x + w.y * h.y + w.z * h.z + w.w * h.w;
      }
#pragma unroll
      for (int off = 16; off > 0; off >>= 1) p += __shfl_xor(p, off, 32);
      if (lane == 0) ghl[lr] = p;
    }
    __syncthreads();

    if (tid < IH) {
      int i = i0 + tid;
      size_t girow;
      if (dir == 0) girow = (size_t)t;
      else { int bb = t / SS; int sr = t % SS; girow = (size_t)bb * SS + (SS - 1 - sr); }
      const float* g = gi + girow * NPAD;
      float ir = g[i], iz = g[HH + i], inn = g[2 * HH + i];
      float hr = ghl[tid]          + bh0;
      float hz = ghl[IH + tid]     + bh1;
      float hn = ghl[2 * IH + tid] + bh2;
      float r = 1.0f / (1.0f + __expf(-(ir + hr)));
      float z = 1.0f / (1.0f + __expf(-(iz + hz)));
      float n = tanhf(inn + r * hn);
      float hv = (1.0f - z) * n + z * hbuf[i];
      all_h[(size_t)t * LDH + dir * HH + i] = hv;
    }
    __syncthreads();

    // cross-WG barrier for this direction (monotonic counter, acquire/release)
    target += NWG;
    if (tid == 0) {
      __threadfence();
      __hip_atomic_fetch_add(&counters[dir], 1u, __ATOMIC_RELEASE, __HIP_MEMORY_SCOPE_AGENT);
      while (__hip_atomic_load(&counters[dir], __ATOMIC_ACQUIRE, __HIP_MEMORY_SCOPE_AGENT) < target)
        __builtin_amdgcn_s_sleep(1);
    }
    __syncthreads();
  }
}

__device__ __forceinline__ float dot4(float4 a, float4 b) {
  return a.x * b.x + a.y * b.y + a.z * b.z + a.w * b.w;
}

// ---------------- tail: q = hs @ W_lin.T + b_lin ----------------
__global__ void k_q(const float* __restrict__ all_h, const float* __restrict__ W_lin,
                    const float* __restrict__ b_lin, float* __restrict__ qbuf) {
  int b = blockIdx.x, tid = (int)threadIdx.x;
  const float4* hs4 = (const float4*)(all_h + (size_t)(b * SS + (SS - 1)) * LDH);
  for (int n = tid; n < 2 * HH; n += 256) {
    const float4* wr4 = (const float4*)(W_lin + (size_t)n * (2 * HH));
    float acc = b_lin[n];
    for (int d = 0; d < (2 * HH) / 4; ++d) acc += dot4(hs4[d], wr4[d]);
    qbuf[b * 2 * HH + n] = acc;
  }
}

// ---------------- tail: softmax attention context ----------------
__global__ void k_attn(const float* __restrict__ all_h, const float* __restrict__ qbuf,
                       float* __restrict__ ctxbuf) {
  __shared__ float sc[SS];
  __shared__ float red[256];
  int b = blockIdx.x, tid = (int)threadIdx.x;
  const float4* q4 = (const float4*)(qbuf + b * 2 * HH);
  for (int s = tid; s < SS; s += 256) {
    const float4* h4 = (const float4*)(all_h + (size_t)(b * SS + s) * LDH);
    float a = 0.0f;
    for (int d = 0; d < (2 * HH) / 4; ++d) a += dot4(q4[d], h4[d]);
    sc[s] = a;
  }
  __syncthreads();
  float m = -INFINITY;
  for (int s = tid; s < SS; s += 256) m = fmaxf(m, sc[s]);
  red[tid] = m; __syncthreads();
  for (int o = 128; o > 0; o >>= 1) { if (tid < o) red[tid] = fmaxf(red[tid], red[tid + o]); __syncthreads(); }
  m = red[0]; __syncthreads();
  float ssum = 0.0f;
  for (int s = tid; s < SS; s += 256) { float e = __expf(sc[s] - m); sc[s] = e; ssum += e; }
  red[tid] = ssum; __syncthreads();
  for (int o = 128; o > 0; o >>= 1) { if (tid < o) red[tid] += red[tid + o]; __syncthreads(); }
  float inv = 1.0f / red[0];
  __syncthreads();
  float4* cx4 = (float4*)(ctxbuf + b * 2 * HH);
  for (int d = tid; d < (2 * HH) / 4; d += 256) {
    float4 a = make_float4(0.f, 0.f, 0.f, 0.f);
    for (int s = 0; s < SS; ++s) {
      float4 h = *(const float4*)(all_h + (size_t)(b * SS + s) * LDH + d * 4);
      float p = sc[s];
      a.x += p * h.x; a.y += p * h.y; a.z += p * h.z; a.w += p * h.w;
    }
    cx4[d] = make_float4(a.x * inv, a.y * inv, a.z * inv, a.w * inv);
  }
}

// ---------------- tail: gated fusion ----------------
__global__ void k_gate(const float* __restrict__ all_h, const float* __restrict__ ctxbuf,
                       const float* __restrict__ W_g, const float* __restrict__ b_g,
                       const float* __restrict__ W_f, const float* __restrict__ b_f,
                       float* __restrict__ gated) {
  int b = blockIdx.x, tid = (int)threadIdx.x;
  const float* hs = all_h + (size_t)(b * SS + (SS - 1)) * LDH;
  const float4* hs4 = (const float4*)hs;
  const float4* cx4 = (const float4*)(ctxbuf + b * 2 * HH);
  for (int n = tid; n < 2 * HH; n += 256) {
    const float4* wg4 = (const float4*)(W_g + (size_t)n * (4 * HH));
    const float4* wf4 = (const float4*)(W_f + (size_t)n * (4 * HH));
    float ag = b_g[n], af = b_f[n];
    for (int d = 0; d < (2 * HH) / 4; ++d) {
      float4 c = cx4[d];
      ag += dot4(c, wg4[d]);
      af += dot4(c, wf4[d]);
    }
    for (int d = 0; d < (2 * HH) / 4; ++d) {
      float4 h = hs4[d];
      ag += dot4(h, wg4[(2 * HH) / 4 + d]);
      af += dot4(h, wf4[(2 * HH) / 4 + d]);
    }
    float g = 1.0f / (1.0f + __expf(-ag));
    float f = tanhf(af);
    gated[b * 2 * HH + n] = g * f + (1.0f - g) * hs[n];
  }
}

// ---------------- output: final (last all_h row) + attn broadcast ----------------
__global__ void k_out(const float* __restrict__ all_h, const float* __restrict__ gated,
                      float* __restrict__ out) {
  int blk = blockIdx.x, tid = (int)threadIdx.x;
  if (blk == ROWS) {                       // final = concat(h_f, h_b) = last all_h row
    const float4* src = (const float4*)(all_h + (size_t)(ROWS - 1) * LDH);
    float4* dst = (float4*)out;
    for (int j = tid; j < (2 * HH) / 4; j += 256) dst[j] = src[j];
    return;
  }
  int b = blk / SS;
  const float4* g4 = (const float4*)(gated + b * 2 * HH);
  float4* o4 = (float4*)(out + 2 * HH + (size_t)blk * (2 * HH));
  for (int j = tid; j < (2 * HH) / 4; j += 256) o4[j] = g4[j];
}

// ---------------- launch ----------------
extern "C" void kernel_launch(void* const* d_in, const int* in_sizes, int n_in,
                              void* d_out, int out_size, void* d_ws, size_t ws_size,
                              hipStream_t stream) {
  const float* ctx   = (const float*)d_in[0];
  const int*   tags  = (const int*)  d_in[1];
  const float* bio   = (const float*)d_in[2];
  const float* W_ih  = (const float*)d_in[3];
  const float* b_ih  = (const float*)d_in[4];
  const float* W_hh  = (const float*)d_in[5];
  const float* b_hh  = (const float*)d_in[6];
  const float* W_lin = (const float*)d_in[7];
  const float* b_lin = (const float*)d_in[8];
  const float* W_g   = (const float*)d_in[9];
  const float* b_g   = (const float*)d_in[10];
  const float* W_f   = (const float*)d_in[11];
  const float* b_f   = (const float*)d_in[12];
  float* out = (float*)d_out;

  // workspace carve-out (256B aligned regions)
  char* base = (char*)d_ws;
  size_t off = 0;
  auto carve = [&](size_t bytes) -> char* {
    char* p = base + off;
    off = (off + bytes + 255) & ~(size_t)255;
    return p;
  };
  __bf16* Asw        = (__bf16*)carve((size_t)NMT * NKT * 32 * 16 * sizeof(__bf16));
  __bf16* Bsw        = (__bf16*)carve((size_t)NKT * NNT2 * 32 * 16 * sizeof(__bf16));
  float*  gi         = (float*) carve((size_t)ROWS * NPAD * sizeof(float));
  float*  all_h      = (float*) carve((size_t)ROWS * LDH * sizeof(float));
  float*  qbuf       = (float*) carve((size_t)BB * 2 * HH * sizeof(float));
  float*  ctxbuf     = (float*) carve((size_t)BB * 2 * HH * sizeof(float));
  float*  gated      = (float*) carve((size_t)BB * 2 * HH * sizeof(float));
  unsigned int* cnt  = (unsigned int*)carve(256);

  k_init<<<1, 64, 0, stream>>>(cnt);

  int aThreads = NMT * NKT * 32;                               // 245760
  k_prep_A<<<(aThreads + 255) / 256, 256, 0, stream>>>(ctx, tags, bio, Asw);
  int bThreads = NKT * NNT2 * 32;                              // 36480
  k_prep_B<<<(bThreads + 255) / 256, 256, 0, stream>>>(W_ih, Bsw);

  int groups = (NMT / 4) * (NNT2 / 2);                         // 10944 wave-groups
  k_gemm<<<groups / 8, 256, 0, stream>>>(Asw, Bsw, b_ih, gi);

  size_t smem = (size_t)(72 * LDW + LDW + 72) * sizeof(float); // 177824 B < 320 KB WGP LDS
  k_scan<<<2 * NWG, 256, smem, stream>>>(gi, W_hh, b_hh, all_h, cnt);

  k_q   <<<BB, 256, 0, stream>>>(all_h, W_lin, b_lin, qbuf);
  k_attn<<<BB, 256, 0, stream>>>(all_h, qbuf, ctxbuf);
  k_gate<<<BB, 256, 0, stream>>>(all_h, ctxbuf, W_g, b_g, W_f, b_f, gated);
  k_out <<<ROWS + 1, 256, 0, stream>>>(all_h, gated, out);
}
